// TransformerBlock_41326175322732
// MI455X (gfx1250) — matrix operand order repaired and checked by reading, not compiled, not run on hardware
//
#include <hip/hip_runtime.h>
#include <math.h>

// ---------------------------------------------------------------------------
// MI455X / gfx1250 transformer block.
// All matmuls use v_wmma_f32_16x16x32_bf16 (fp32 accumulate). fp32 operands
// are converted to bf16 on the fly while filling LDS tiles; the 192MB L2
// absorbs repeated fp32 weight reads. The [4096,4096] MLP hidden tensor is
// kept bf16 and its GEMM A-tiles are moved HBM->LDS with
// global_load_async_to_lds_b128 (ASYNCcnt). GELU is branch-free via native
// v_exp_f32 + v_rcp_f32. Attention is flash-style, one wave per
// (batch, head, 16-query tile), fully on the WMMA pipe.
// ---------------------------------------------------------------------------

typedef __bf16 bf16;
typedef __attribute__((ext_vector_type(16))) __bf16 v16bf;
typedef __attribute__((ext_vector_type(8)))  __bf16 v8bf;
typedef __attribute__((ext_vector_type(8)))  float  v8f;

#define EMB   1024
#define HID   4096
#define SEQ   1024
#define BATCH 4
#define HEADS 16
#define DH    64
#define ROWS  (BATCH * SEQ)   // 4096 token rows

// Branch-free tanh-GELU: 0.5x(1+tanh(u)) = x * sigmoid(2u) = x / (1 + e^-2u)
__device__ __forceinline__ float gelu_f(float x) {
  const float u = 0.7978845608028654f * (x + 0.044715f * x * x * x);
  const float e = __expf(-2.0f * u);                 // native v_exp_f32
  return x * __builtin_amdgcn_rcpf(1.0f + e);        // native v_rcp_f32
}

__device__ __forceinline__ v16bf cat16(v8bf lo, v8bf hi) {
  return __builtin_shufflevector(lo, hi, 0, 1, 2, 3, 4, 5, 6, 7,
                                         8, 9, 10, 11, 12, 13, 14, 15);
}

union BF16x16 { v16bf v; bf16 e[16]; };

// Low 32 bits of a generic pointer to __shared__ = wave-relative LDS offset
// (LDS aperture lives entirely in addr[63:32] on CDNA5).
__device__ __forceinline__ unsigned int lds_addr32(const void* p) {
  return (unsigned int)(unsigned long long)p;
}

// ---------------------------------------------------------------------------
// GEMM: C[M,N] = act(A[M,K] @ W[K,N] + bias)
// ABF: A is bf16 (1, async HBM->LDS) or fp32 (0, convert at fill).
// ACT: 1 -> gelu.  OBF: C bf16 (1) or fp32 (0).
// Block 256 thr (8 waves). Tile 128x128x32. Wave: 32 rows x 64 cols
// = 2x4 v_wmma_f32_16x16x32_bf16.
// ---------------------------------------------------------------------------
template<int ABF, int ACT, int OBF>
__global__ __launch_bounds__(256)
void gemm_bf16_wmma(const void* __restrict__ Ap, const float* __restrict__ W,
                    const float* __restrict__ bias, void* __restrict__ Cp,
                    int M, int N, int K)
{
  (void)M;
  constexpr int LDT = 40;                 // 80B row stride: 16B aligned, bank-spread
  __shared__ bf16 As[128 * LDT];
  __shared__ bf16 Bs[128 * LDT];

  const int tid  = threadIdx.x;
  const int lane = tid & 31;
  const int wave = tid >> 5;
  const int wm   = wave & 3;              // 4 row-groups of 32
  const int wn   = wave >> 2;             // 2 col-groups of 64
  const int bm   = blockIdx.y * 128;
  const int bn   = blockIdx.x * 128;

  v8f acc[2][4];
#pragma unroll
  for (int i = 0; i < 2; ++i)
#pragma unroll
    for (int j = 0; j < 4; ++j) acc[i][j] = v8f{0.f,0.f,0.f,0.f,0.f,0.f,0.f,0.f};

  const int arow  = lane & 15;
  const int ahalf = (lane >> 4) << 3;     // unused; A frag uses bkoff (0 / 16)
  const int bkoff = (lane >> 4) << 4;     // B frag K-half: 0 / 16

  // per-thread tile-fill coordinates
  const int ar = tid >> 1;                // A: row 0..127
  const int ac = (tid & 1) << 4;          // A: 16-col half
  const int kl = tid >> 3;                // W: k row 0..31
  const int ch = tid & 7;                 // W: 16-col chunk 0..7

  for (int k0 = 0; k0 < K; k0 += 32) {
    { // ---- A tile 128x32 (8KB bf16)
      if (ABF) {
        // bf16 source: async DMA straight into LDS, no VGPR round-trip
        const unsigned long long ga =
            (unsigned long long)((const char*)Ap +
                ((size_t)(bm + ar) * K + k0 + ac) * sizeof(bf16));
        const unsigned int la = lds_addr32(&As[ar * LDT + ac]);
        asm volatile("global_load_async_to_lds_b128 %0, %1, off"
                     :: "v"(la), "v"(ga) : "memory");
        asm volatile("global_load_async_to_lds_b128 %0, %1, off offset:16"
                     :: "v"(la), "v"(ga) : "memory");
        if (k0 + 32 < K)
          __builtin_prefetch((const char*)Ap +
              ((size_t)(bm + ar) * K + k0 + 32 + ac) * sizeof(bf16), 0, 3);
      } else {
        const float* src = (const float*)Ap + (size_t)(bm + ar) * K + k0 + ac;
        bf16* dst = &As[ar * LDT + ac];
#pragma unroll
        for (int i = 0; i < 16; ++i) dst[i] = (bf16)src[i];
        if (k0 + 32 < K) __builtin_prefetch(src + 32, 0, 3);
      }
    }
    { // ---- W tile 32x128, stored transposed: Bs[n][k] (fp32 -> bf16)
      const float* src = W + (size_t)(k0 + kl) * N + bn + (ch << 4);
#pragma unroll
      for (int i = 0; i < 16; ++i) Bs[(ch * 16 + i) * LDT + kl] = (bf16)src[i];
      if (k0 + 32 < K) __builtin_prefetch(src + (size_t)32 * N, 0, 3);
    }
    if (ABF) asm volatile("s_wait_asynccnt 0x0" ::: "memory");
    __syncthreads();

    v16bf a[2];
#pragma unroll
    for (int i = 0; i < 2; ++i) {         // A 16x32: lanes<16 K 0-15, >=16 K 16-31
      const bf16* p = &As[(wm * 32 + i * 16 + arow) * LDT];
      a[i] = cat16(*(const v8bf*)(p + bkoff), *(const v8bf*)(p + bkoff + 8));
    }
#pragma unroll
    for (int j = 0; j < 4; ++j) {         // B 32x16: lanes<16 K 0-15, >=16 K 16-31
      const bf16* p = &Bs[(wn * 64 + j * 16 + (lane & 15)) * LDT + bkoff];
      v16bf b = cat16(*(const v8bf*)p, *(const v8bf*)(p + 8));
#pragma unroll
      for (int i = 0; i < 2; ++i)
        acc[i][j] = __builtin_amdgcn_wmma_f32_16x16x32_bf16(
            false, a[i], false, b, (short)0, acc[i][j], false, false);
    }
    __syncthreads();
  }

  // C layout: lanes<16 rows r, lanes>=16 rows 8+r; N = lane&15
  const int rbase = bm + wm * 32 + ((lane >> 4) << 3);
  const int lcol  = lane & 15;
#pragma unroll
  for (int j = 0; j < 4; ++j) {
    const int gcol = bn + wn * 64 + j * 16 + lcol;
    const float bv = bias[gcol];
#pragma unroll
    for (int i = 0; i < 2; ++i) {
#pragma unroll
      for (int r = 0; r < 8; ++r) {
        float v = acc[i][j][r] + bv;
        if (ACT) v = gelu_f(v);
        const size_t idx = (size_t)(rbase + i * 16 + r) * N + gcol;
        if (OBF) ((bf16*)Cp)[idx] = (bf16)v;
        else     ((float*)Cp)[idx] = v;
      }
    }
  }
}

// ---------------------------------------------------------------------------
// Flash-style attention: one wave = one (batch, head, 16-query tile).
// 32 keys per online-softmax step; scores and P@V both on WMMA bf16.
// ---------------------------------------------------------------------------
template<int CAUSAL>
__global__ __launch_bounds__(256)
void attn_wmma(const float* __restrict__ Q, const float* __restrict__ Km,
               const float* __restrict__ Vm, float* __restrict__ Om)
{
  __shared__ bf16 Pbuf[8][16 * 32];       // per-wave P staging (C->A layout fixup)
  const int lane = threadIdx.x & 31;
  const int wave = threadIdx.x >> 5;
  const int id = blockIdx.x * 8 + wave;   // 0..4095
  const int qt = id & 63;
  const int h  = (id >> 6) & 15;
  const int n  = id >> 10;

  const float* qb = Q + ((size_t)n * SEQ + qt * 16) * EMB + h * DH;
  const float* kb = Km + (size_t)n * SEQ * EMB + h * DH;
  const float* vb = Vm + (size_t)n * SEQ * EMB + h * DH;

  const int arow  = lane & 15;
  const int ahalf = (lane >> 4) << 3;
  const int bkoff = (lane >> 4) << 4;
  const int rbase = (lane >> 4) << 3;

  // Q fragments (pre-scaled by 1/sqrt(64)), dh chunks 0-31 / 32-63
  v16bf qa[2];
#pragma unroll
  for (int c = 0; c < 2; ++c) {
    const float* p = qb + (size_t)arow * EMB + c * 32;
    BF16x16 f;
#pragma unroll
    for (int i = 0; i < 8; ++i) f.e[i]     = (bf16)(p[bkoff + i] * 0.125f);
#pragma unroll
    for (int i = 0; i < 8; ++i) f.e[8 + i] = (bf16)(p[bkoff + 8 + i] * 0.125f);
    qa[c] = f.v;
  }

  v8f o[4];
#pragma unroll
  for (int t = 0; t < 4; ++t) o[t] = v8f{0.f,0.f,0.f,0.f,0.f,0.f,0.f,0.f};
  float mrow[8], lrow[8];
#pragma unroll
  for (int r = 0; r < 8; ++r) { mrow[r] = -1e30f; lrow[r] = 0.0f; }

  const int niter = CAUSAL ? ((qt >> 1) + 1) : (SEQ / 32);
  for (int it = 0; it < niter; ++it) {
    const int j0 = it * 32;
    // ---- scores: two 16x16 tiles (keys j0..j0+31)
    v8f s[2];
#pragma unroll
    for (int t = 0; t < 2; ++t) {
      s[t] = v8f{0.f,0.f,0.f,0.f,0.f,0.f,0.f,0.f};
#pragma unroll
      for (int c = 0; c < 2; ++c) {
        // B = K^T: column = key, K-dim = dh -> contiguous 16-float run per lane
        const float* p = kb + (size_t)(j0 + t * 16 + (lane & 15)) * EMB + c * 32 + bkoff;
        BF16x16 f;
#pragma unroll
        for (int i = 0; i < 16; ++i) f.e[i] = (bf16)p[i];
        s[t] = __builtin_amdgcn_wmma_f32_16x16x32_bf16(
            false, qa[c], false, f.v, (short)0, s[t], false, false);
      }
    }
    if (CAUSAL) {
      const int qrow0 = qt * 16 + rbase;
#pragma unroll
      for (int t = 0; t < 2; ++t) {
        const int kg = j0 + t * 16 + (lane & 15);
#pragma unroll
        for (int r = 0; r < 8; ++r)
          s[t][r] = (kg > qrow0 + r) ? -1e30f : s[t][r];
      }
    }
    // ---- online softmax (row reductions within 16-lane halves)
    float mnew[8], alpha[8];
#pragma unroll
    for (int r = 0; r < 8; ++r) {
      float v = fmaxf(s[0][r], s[1][r]);
#pragma unroll
      for (int m = 8; m >= 1; m >>= 1) v = fmaxf(v, __shfl_xor(v, m, 32));
      mnew[r]  = fmaxf(mrow[r], v);
      alpha[r] = __expf(mrow[r] - mnew[r]);
      mrow[r]  = mnew[r];
    }
#pragma unroll
    for (int t = 0; t < 2; ++t)
#pragma unroll
      for (int r = 0; r < 8; ++r) s[t][r] = __expf(s[t][r] - mnew[r]);
#pragma unroll
    for (int r = 0; r < 8; ++r) {
      float v = s[0][r] + s[1][r];
#pragma unroll
      for (int m = 8; m >= 1; m >>= 1) v += __shfl_xor(v, m, 32);
      lrow[r] = lrow[r] * alpha[r] + v;
    }
#pragma unroll
    for (int t = 0; t < 4; ++t)
#pragma unroll
      for (int r = 0; r < 8; ++r) o[t][r] *= alpha[r];

    // ---- P (16x32) through LDS: C-layout -> A-fragment layout
    bf16* pb = Pbuf[wave];
#pragma unroll
    for (int t = 0; t < 2; ++t)
#pragma unroll
      for (int r = 0; r < 8; ++r)
        pb[(rbase + r) * 32 + t * 16 + (lane & 15)] = (bf16)s[t][r];
    asm volatile("s_wait_dscnt 0x0" ::: "memory");   // cross-lane RAW through LDS
    v16bf pA;
    {
      const bf16* p = &pb[(lane & 15) * 32];
      pA = cat16(*(const v8bf*)(p + bkoff), *(const v8bf*)(p + bkoff + 8));
    }
    // ---- O += P @ V (4 dh tiles of 16)
#pragma unroll
    for (int t = 0; t < 4; ++t) {
      BF16x16 f;
      const int dh = t * 16 + (lane & 15);
#pragma unroll
      for (int i = 0; i < 16; ++i)
        f.e[i] = (bf16)vb[(size_t)(j0 + bkoff + i) * EMB + dh];
      o[t] = __builtin_amdgcn_wmma_f32_16x16x32_bf16(
          false, pA, false, f.v, (short)0, o[t], false, false);
    }
  }

  float* ob = Om + ((size_t)n * SEQ + qt * 16) * EMB + h * DH;
  float rl[8];
#pragma unroll
  for (int r = 0; r < 8; ++r) rl[r] = __builtin_amdgcn_rcpf(lrow[r]);
#pragma unroll
  for (int t = 0; t < 4; ++t)
#pragma unroll
    for (int r = 0; r < 8; ++r)
      ob[(size_t)(rbase + r) * EMB + t * 16 + (lane & 15)] = o[t][r] * rl[r];
}

// ---------------------------------------------------------------------------
// LayerNorm (row = 1024). ADDG=1: t = gelu(x + skip) first.
// ---------------------------------------------------------------------------
__device__ __forceinline__ float block_reduce_sum(float v, float* red) {
  const int lane = threadIdx.x & 31;
  const int wv   = threadIdx.x >> 5;
#pragma unroll
  for (int m = 16; m >= 1; m >>= 1) v += __shfl_xor(v, m, 32);
  if (lane == 0) red[wv] = v;
  __syncthreads();
  float t = (threadIdx.x < 8) ? red[threadIdx.x] : 0.0f;
  if (wv == 0) {
#pragma unroll
    for (int m = 4; m >= 1; m >>= 1) t += __shfl_xor(t, m, 32);
    if (lane == 0) red[0] = t;
  }
  __syncthreads();
  float r = red[0];
  __syncthreads();
  return r;
}

template<int ADDG>
__global__ __launch_bounds__(256)
void ln_kernel(const float* __restrict__ X, const float* __restrict__ S,
               const float* __restrict__ g, const float* __restrict__ b,
               float* __restrict__ out)
{
  __shared__ float red[8];
  const int row = blockIdx.x;
  const float* xr = X + (size_t)row * EMB;
  float v[4];
  float s = 0.0f;
#pragma unroll
  for (int i = 0; i < 4; ++i) {
    const int c = threadIdx.x + i * 256;
    float t = xr[c];
    if (ADDG) t = gelu_f(t + S[(size_t)row * EMB + c]);
    v[i] = t;
    s += t;
  }
  const float mean = block_reduce_sum(s, red) * (1.0f / EMB);
  float s2 = 0.0f;
#pragma unroll
  for (int i = 0; i < 4; ++i) { float d = v[i] - mean; s2 += d * d; }
  const float var  = block_reduce_sum(s2, red) * (1.0f / EMB);
  const float rstd = rsqrtf(var + 1e-5f);
#pragma unroll
  for (int i = 0; i < 4; ++i) {
    const int c = threadIdx.x + i * 256;
    out[(size_t)row * EMB + c] = (v[i] - mean) * rstd * g[c] + b[c];
  }
}

// ---------------------------------------------------------------------------
// Host orchestration.
// Input order (setup_inputs dict, nested dicts flattened in insertion order):
//  0 x, 1 enc,
//  2-5 sa_q{W1,b1,W2,b2}, 6-9 sa_k, 10-13 sa_v, 14-15 sa_res{W,b}, 16-17 sa_ln{g,b},
//  18-21 ca_q, 22-25 ca_k, 26-29 ca_v, 30-31 ca_res, 32-33 ca_ln,
//  34-37 ff, 38-39 ff_inner_ln, 40-41 ff_res, 42-43 ff_ln
// ---------------------------------------------------------------------------
extern "C" void kernel_launch(void* const* d_in, const int* in_sizes, int n_in,
                              void* d_out, int out_size, void* d_ws, size_t ws_size,
                              hipStream_t stream)
{
  (void)in_sizes; (void)n_in; (void)out_size; (void)ws_size;
  const float* x   = (const float*)d_in[0];
  const float* enc = (const float*)d_in[1];
  auto F = [&](int i) { return (const float*)d_in[i]; };

  // workspace carve: 6 x 16MiB fp32 + 32MiB bf16 hidden = 128 MiB
  char* w = (char*)d_ws;
  const size_t FB = (size_t)ROWS * EMB * sizeof(float);
  float* qb   = (float*)(w + 0 * FB);
  float* kbuf = (float*)(w + 1 * FB);   // k; reused as residual-gemm out
  float* vbuf = (float*)(w + 2 * FB);   // v; reused as ff-inner-LN out
  float* attn = (float*)(w + 3 * FB);   // attn out; reused as ff two_layer out
  float* xa   = (float*)(w + 4 * FB);   // x after SA block
  float* xb   = (float*)(w + 5 * FB);   // x after CA block
  bf16*  hbf  = (bf16*)(w + 6 * FB);    // [4096,4096] bf16 MLP hidden

  dim3 blk(256);
  dim3 gHID(HID / 128, ROWS / 128);     // N=4096
  dim3 gEMB(EMB / 128, ROWS / 128);     // N=1024
  const int attnBlocks = BATCH * HEADS * (SEQ / 16) / 8;  // 512

  auto two_layer = [&](const float* in, int p, float* out, bool act2) {
    // h = gelu(in @ W1 + b1) -> bf16 hidden ; out = [gelu](h @ W2 + b2)
    gemm_bf16_wmma<0, 1, 1><<<gHID, blk, 0, stream>>>(in, F(p), F(p + 1), hbf, ROWS, HID, EMB);
    if (act2)
      gemm_bf16_wmma<1, 1, 0><<<gEMB, blk, 0, stream>>>(hbf, F(p + 2), F(p + 3), out, ROWS, EMB, HID);
    else
      gemm_bf16_wmma<1, 0, 0><<<gEMB, blk, 0, stream>>>(hbf, F(p + 2), F(p + 3), out, ROWS, EMB, HID);
  };

  // ---- self attention (causal); v block has gelu on layer 2
  two_layer(x, 2, qb, false);
  two_layer(x, 6, kbuf, false);
  two_layer(x, 10, vbuf, true);
  attn_wmma<1><<<attnBlocks, blk, 0, stream>>>(qb, kbuf, vbuf, attn);
  gemm_bf16_wmma<0, 1, 0><<<gEMB, blk, 0, stream>>>(attn, F(14), F(15), kbuf, ROWS, EMB, EMB);
  ln_kernel<1><<<ROWS, blk, 0, stream>>>(kbuf, x, F(16), F(17), xa);

  // ---- cross attention (no mask)
  two_layer(xa, 18, qb, false);
  two_layer(enc, 22, kbuf, false);
  two_layer(enc, 26, vbuf, false);
  attn_wmma<0><<<attnBlocks, blk, 0, stream>>>(qb, kbuf, vbuf, attn);
  gemm_bf16_wmma<0, 1, 0><<<gEMB, blk, 0, stream>>>(attn, F(30), F(31), kbuf, ROWS, EMB, EMB);
  ln_kernel<1><<<ROWS, blk, 0, stream>>>(kbuf, xa, F(32), F(33), xb);

  // ---- feed-forward: gelu both layers, inner LN, residual, final LN
  two_layer(xb, 34, attn, true);
  ln_kernel<0><<<ROWS, blk, 0, stream>>>(attn, nullptr, F(38), F(39), vbuf);
  gemm_bf16_wmma<0, 1, 0><<<gEMB, blk, 0, stream>>>(vbuf, F(40), F(41), kbuf, ROWS, EMB, EMB);
  ln_kernel<1><<<ROWS, blk, 0, stream>>>(kbuf, xb, F(42), F(43), (float*)d_out);
}
